// F0Resonance_61040075211187
// MI455X (gfx1250) — matrix operand order repaired and checked by reading, not compiled, not run on hardware
//
#include <hip/hip_runtime.h>

typedef __attribute__((ext_vector_type(16))) _Float16 v16h;
typedef __attribute__((ext_vector_type(8)))  float    v8f;

#define N_OCT        32
#define N_SAMPLES_K  32768
#define TILES_PER_BE 128            // 32768 / 256 samples per 16x16 tile
#define WAVES_PER_BLK 8

__global__ __launch_bounds__(256)
void F0Resonance_osc_wmma(const float* __restrict__ f0,
                          const float* __restrict__ dc,
                          const float* __restrict__ fs,
                          float* __restrict__ out)
{
    __shared__ float s_f[N_OCT];
    __shared__ float s_dec[N_OCT];

    const int be  = blockIdx.x;          // 0..63  (B*E rows)
    const int tid = threadIdx.x;
    const int lane = tid & 31;
    const int wave = tid >> 5;

    // ---- per-row setup: f[o] and decay[o] (wave 0, lanes 0..31, uniform per wave)
    if (tid < N_OCT) {
        const float min_f = 20.0f   / 11025.0f;
        const float max_f = 3000.0f / 11025.0f;
        const float pi    = 3.14159265358979323846f;

        float f0v = fabsf(f0[be]);
        float fv  = (min_f + f0v * (max_f - min_f)) * pi;
        float f   = fv * fs[be] * (float)(tid + 1);      // cumsum of const spacing
        if (!(f < 1.0f)) f = 0.0f;                       // Nyquist mask -> zero contribution

        float x  = dc[be];
        float s1 = 1.0f / (1.0f + __expf(-x));           // double sigmoid (matches ref)
        float s2 = 1.0f / (1.0f + __expf(-s1));
        float d  = 0.01f + s2 * 0.9801f;                 // BASE + dv*(1-BASE)*0.99
        float dec = __expf((float)(tid + 1) * __logf(d + 1e-12f)); // d^(o+1)

        s_f[tid]   = f;
        s_dec[tid] = dec;
    }
    __syncthreads();

    // ---- B matrix (64x16 logical, split into two K=32 operands), constant per row.
    // B1[k][r] = cos(f[k]*(r+1)), B2[k][r] = sin(f[k]*(r+1))
    // B K order chosen here (matches A): lanes 0-15 hold K=0..15 (elem e -> K=e), lanes 16-31 K=16..31; N = lane&15.
    v16h b1, b2;
    {
        const int r     = lane & 15;
        const int kbase = (lane >> 4) << 4;
        #pragma unroll
        for (int e = 0; e < 16; ++e) {
            float f  = s_f[kbase + e];
            float sv, cv;
            __sincosf(f * (float)(r + 1), &sv, &cv);
            b1[e] = (_Float16)cv;
            b2[e] = (_Float16)sv;
        }
    }

    const int m = lane & 15;                 // A-matrix row this lane supplies
    float* const obase = out + (size_t)be * N_SAMPLES_K;

    // ---- each wave processes 16 of the 128 tiles of this row
    for (int t = wave; t < TILES_PER_BE; t += WAVES_PER_BLK) {
        const int   q0 = t * 16;
        const float qf = 16.0f * (float)(q0 + m);        // block-start phase multiplier

        // A1[m][o] = decay[o]*sin(16*q*f[o]),  A2[m][o] = decay[o]*cos(16*q*f[o])
        // A K order chosen here (matches B): M = lane&15; elem e -> K = e + (lane>>4)*16
        v16h a1, a2;
        #pragma unroll
        for (int e = 0; e < 16; ++e) {
            int   k   = e + ((lane >> 4) << 4);
            float f   = s_f[k];
            float dec = s_dec[k];
            float sv, cv;
            __sincosf(f * qf, &sv, &cv);
            a1[e] = (_Float16)(dec * sv);
            a2[e] = (_Float16)(dec * cv);
        }

        v8f c = {};
        c = __builtin_amdgcn_wmma_f32_16x16x32_f16(false, a1, false, b1,
                                                   (short)0, c, false, false);
        c = __builtin_amdgcn_wmma_f32_16x16x32_f16(false, a2, false, b2,
                                                   (short)0, c, false, false);

        // D layout: elem v -> M = v + (lane>>4)*8, N = lane&15 ; sample = (q0+M)*16 + N
        const int n    = lane & 15;
        const int mrow = (lane >> 4) << 3;
        #pragma unroll
        for (int v = 0; v < 8; ++v) {
            obase[(q0 + mrow + v) * 16 + n] = c[v];
        }
    }
}

__global__ __launch_bounds__(1024)
void F0Resonance_normalize(float* __restrict__ out)
{
    __shared__ float smax[32];

    const int be  = blockIdx.x;
    const int tid = threadIdx.x;
    float* const row = out + (size_t)be * N_SAMPLES_K;

    float vals[32];
    float mx = 0.0f;
    #pragma unroll
    for (int i = 0; i < 32; ++i) {
        vals[i] = row[tid + i * 1024];
        mx = fmaxf(mx, fabsf(vals[i]));
    }
    // intra-wave reduction (wave32)
    #pragma unroll
    for (int off = 16; off > 0; off >>= 1)
        mx = fmaxf(mx, __shfl_xor(mx, off, 32));
    if ((tid & 31) == 0) smax[tid >> 5] = mx;
    __syncthreads();
    if (tid < 32) {
        float m2 = smax[tid];
        #pragma unroll
        for (int off = 16; off > 0; off >>= 1)
            m2 = fmaxf(m2, __shfl_xor(m2, off, 32));
        smax[tid] = m2;
    }
    __syncthreads();

    const float inv = 1.0f / (smax[0] + 1e-8f);
    #pragma unroll
    for (int i = 0; i < 32; ++i)
        row[tid + i * 1024] = vals[i] * inv;
}

extern "C" void kernel_launch(void* const* d_in, const int* in_sizes, int n_in,
                              void* d_out, int out_size, void* d_ws, size_t ws_size,
                              hipStream_t stream)
{
    (void)in_sizes; (void)n_in; (void)out_size; (void)d_ws; (void)ws_size;

    const float* f0 = (const float*)d_in[0];   // (B,E,1) -> 64
    const float* dc = (const float*)d_in[1];   // decay_coefficients
    const float* fs = (const float*)d_in[2];   // freq_spacing
    float* out = (float*)d_out;                // (B,E,S) -> 64 * 32768

    F0Resonance_osc_wmma<<<64, 256, 0, stream>>>(f0, dc, fs, out);
    F0Resonance_normalize<<<64, 1024, 0, stream>>>(out);
}
